// MultiHeadSelfAttention_48550310314055
// MI455X (gfx1250) — matrix operand order repaired and checked by reading, not compiled, not run on hardware
//
#include <hip/hip_runtime.h>

// ---------------------------------------------------------------------------
// MI455X (gfx1250) multi-head self-attention, bf16 WMMA pipeline.
// Stages: cast/transpose -> QKV GEMM -> flash attention (ghost softmax,
// causal) -> output projection.  All matrix math via v_wmma_f32_16x16x32_bf16.
// Round 3: attention key-block loop is now software-pipelined (double-
// buffered K fragments, loads for block i+1 issued under block i's softmax
// and P*V), matching the double-buffered GEMM K-loops.
// ---------------------------------------------------------------------------

typedef __attribute__((ext_vector_type(16))) __bf16 v16bf;
typedef __attribute__((ext_vector_type(8)))  __bf16 v8bf;
typedef __attribute__((ext_vector_type(8)))  float  v8f;

#define B_   2
#define S_   2048
#define E_   1024
#define H_   16
#define D_   64
#define BS_  (B_ * S_)          // 4096
#define QKVN (3 * H_ * D_)      // 3072

__device__ __forceinline__ v8f wmma_bf16(v16bf a, v16bf b, v8f c) {
  // D = A(16x32 bf16) * B(32x16 bf16) + C(16x16 f32)
  return __builtin_amdgcn_wmma_f32_16x16x32_bf16(
      /*neg_a=*/false, a, /*neg_b=*/false, b,
      /*c_mod=*/(short)0, c, /*reuse_a=*/false, /*reuse_b=*/false);
}

// A-operand fragment (16-bit, 16x32), contiguous k order:
//   lanes 0-15 (hi=0): elems 0..15 -> K=0..15
//   lanes 16-31(hi=1): elems 0..15 -> K=16..31
__device__ __forceinline__ v16bf load_a_frag(const __bf16* rowptr, int kbase, int hi) {
  const v8bf lo = *reinterpret_cast<const v8bf*>(rowptr + kbase + 16 * hi);
  const v8bf up = *reinterpret_cast<const v8bf*>(rowptr + kbase + 16 * hi + 8);
  v16bf f;
#pragma unroll
  for (int e = 0; e < 8; ++e) { f[e] = lo[e]; f[e + 8] = up[e]; }
  return f;
}

// ---------------------------------------------------------------------------
// Stage 0a: fp32 -> bf16 straight cast.
// ---------------------------------------------------------------------------
__global__ __launch_bounds__(256) void cast_bf16(const float* __restrict__ in,
                                                 __bf16* __restrict__ out, int n) {
  int i = blockIdx.x * 256 + threadIdx.x;
  if (i < n) out[i] = (__bf16)in[i];
}

// ---------------------------------------------------------------------------
// Stage 0b: fp32 [R][C] -> bf16 [C][R] transpose, LDS-tiled 32x32 so both the
// global read and the global write are coalesced.  Block = (32, 8).
// ---------------------------------------------------------------------------
__global__ __launch_bounds__(256) void transpose_bf16(const float* __restrict__ in,
                                                      __bf16* __restrict__ out,
                                                      int R, int C) {
  __shared__ __bf16 tile[32][33];             // +1 pad kills bank conflicts
  const int tx = threadIdx.x, ty = threadIdx.y;
  const int r0 = blockIdx.y * 32, c0 = blockIdx.x * 32;
#pragma unroll
  for (int i = 0; i < 4; ++i) {
    const int r = r0 + ty + i * 8;
    tile[ty + i * 8][tx] = (__bf16)in[(size_t)r * C + c0 + tx];
  }
  __syncthreads();
#pragma unroll
  for (int i = 0; i < 4; ++i) {
    const int c = c0 + ty + i * 8;
    out[(size_t)c * R + r0 + tx] = tile[tx][ty + i * 8];
  }
}

// ---------------------------------------------------------------------------
// Shared GEMM machinery: wave tile 32(M) x 64(N), K-step 32, double-buffered.
// Block = 256 threads = 8 waves arranged 2(M) x 4(N): block tile 64 x 256.
// ---------------------------------------------------------------------------
struct GemmCtx {
  const __bf16* arow0;
  const __bf16* arow1;
  const __bf16* brow[4];
  int hi;
};

__device__ __forceinline__ void gemm_load(const GemmCtx& g, int k,
                                          v16bf (&A)[2], v16bf (&B)[4]) {
  A[0] = load_a_frag(g.arow0, k, g.hi);
  A[1] = load_a_frag(g.arow1, k, g.hi);
#pragma unroll
  for (int t = 0; t < 4; ++t)
    B[t] = *reinterpret_cast<const v16bf*>(g.brow[t] + k);
}

__device__ __forceinline__ void gemm_compute(const v16bf (&A)[2], const v16bf (&B)[4],
                                             v8f (&acc)[2][4]) {
#pragma unroll
  for (int mi = 0; mi < 2; ++mi)
#pragma unroll
    for (int t = 0; t < 4; ++t)
      acc[mi][t] = wmma_bf16(A[mi], B[t], acc[mi][t]);
}

// Runs the whole double-buffered K loop; acc must be zero-initialized.
__device__ __forceinline__ void gemm_kloop(const GemmCtx& g, v8f (&acc)[2][4]) {
  v16bf A0[2], B0[4], A1[2], B1[4];
  gemm_load(g, 0, A0, B0);
  for (int k = 0; k < E_; k += 64) {
    gemm_load(g, k + 32, A1, B1);             // in flight during compute(A0,B0)
    __builtin_prefetch(g.arow0 + k + 128, 0, 3);
    __builtin_prefetch(g.brow[0] + k + 128, 0, 3);
    gemm_compute(A0, B0, acc);
    if (k + 64 < E_) gemm_load(g, k + 64, A0, B0);  // in flight during (A1,B1)
    gemm_compute(A1, B1, acc);
  }
}

__device__ __forceinline__ GemmCtx gemm_setup(const __bf16* Abase, const __bf16* Wt,
                                              int mbase, int nbase, int ln, int hi) {
  GemmCtx g;
  g.hi = hi;
  g.arow0 = Abase + (size_t)(mbase + ln) * E_;
  g.arow1 = Abase + (size_t)(mbase + 16 + ln) * E_;
#pragma unroll
  for (int t = 0; t < 4; ++t)
    g.brow[t] = Wt + (size_t)(nbase + t * 16 + ln) * E_ + 16 * hi;
  return g;
}

// ---------------------------------------------------------------------------
// Stage 1: QKV projection.  X[BS][E] * WqkvT[N][E] -> scatter to
//   Q[b][h][s][d], K[b][h][s][d], Vt[b][h][d][s]   (all bf16)
// ---------------------------------------------------------------------------
__global__ __launch_bounds__(256) void gemm_qkv(const __bf16* __restrict__ X,
                                                const __bf16* __restrict__ Wt,
                                                __bf16* __restrict__ Qb,
                                                __bf16* __restrict__ Kb,
                                                __bf16* __restrict__ Vt) {
  const int lane = threadIdx.x & 31, wave = threadIdx.x >> 5;
  const int hi = lane >> 4, ln = lane & 15;
  const int wm = wave >> 2, wn = wave & 3;
  const int mbase = blockIdx.y * 64 + wm * 32;
  const int nbase = blockIdx.x * 256 + wn * 64;

  GemmCtx g = gemm_setup(X, Wt, mbase, nbase, ln, hi);
  v8f acc[2][4] = {};
  gemm_kloop(g, acc);

  // C element (mi, t, r): row = mbase + mi*16 + r + 8*hi, col = nbase+t*16+ln.
  // A 16-wide tile never crosses a head (64) or q/k/v (1024) boundary.
#pragma unroll
  for (int t = 0; t < 4; ++t) {
    const int c0  = nbase + t * 16;
    const int sel = c0 >> 10;                 // 0=Q, 1=K, 2=V
    const int h   = (c0 & 1023) >> 6;
    const int d   = (c0 & 63) + ln;
#pragma unroll
    for (int mi = 0; mi < 2; ++mi)
#pragma unroll
      for (int r = 0; r < 8; ++r) {
        const int row = mbase + mi * 16 + r + 8 * hi;
        const int b   = row >> 11;            // /S_
        const int s   = row & (S_ - 1);
        const int bh  = b * H_ + h;
        const __bf16 v = (__bf16)acc[mi][t][r];
        if (sel == 0)      Qb[((size_t)bh * S_ + s) * D_ + d] = v;
        else if (sel == 1) Kb[((size_t)bh * S_ + s) * D_ + d] = v;
        else               Vt[((size_t)bh * D_ + d) * S_ + s] = v;
      }
  }
}

// ---------------------------------------------------------------------------
// Stage 2: causal flash attention with ghost softmax (denominator + 1).
// Grid: (B*H, S/64).  Block = 128 threads = 4 waves; wave w owns queries
// [qb, qb+16) x D=64.  Key blocks of 64; 2 K-steps of 32 over D.
// Key-block loop is double-buffered: block i+1's K fragments are loaded
// while block i's softmax + P*V execute.  V fragments are issued right
// after the score WMMAs so the softmax VALU work hides them.
// ---------------------------------------------------------------------------
__global__ __launch_bounds__(128) void attn_fwd(const __bf16* __restrict__ Qb,
                                                const __bf16* __restrict__ Kb,
                                                const __bf16* __restrict__ Vt,
                                                __bf16* __restrict__ Abf) {
  const int lane = threadIdx.x & 31, wave = threadIdx.x >> 5;
  const int hi = lane >> 4, ln = lane & 15;
  const int bh = blockIdx.x;
  const int qb = blockIdx.y * 64 + wave * 16;

  const __bf16* Qh = Qb + (size_t)bh * S_ * D_;
  const __bf16* Kh = Kb + (size_t)bh * S_ * D_;
  const __bf16* Vh = Vt + (size_t)bh * D_ * S_;

  __shared__ __align__(16) __bf16 Plds[4][16][64];   // private strip per wave

  v16bf aq[2];
  {
    const __bf16* qrow = Qh + (size_t)(qb + ln) * D_;
#pragma unroll
    for (int ks = 0; ks < 2; ++ks) aq[ks] = load_a_frag(qrow, ks * 32, hi);
  }

  float m[8], l[8];
#pragma unroll
  for (int r = 0; r < 8; ++r) { m[r] = -3.0e38f; l[r] = 0.0f; }
  v8f O[4] = {};

  const int kend = qb + 15;                  // causal bound for this wave

  auto load_bk = [&](int kb0, v16bf (&bk)[4][2]) {
#pragma unroll
    for (int t = 0; t < 4; ++t) {
      const __bf16* kp = Kh + (size_t)(kb0 + t * 16 + ln) * D_;
#pragma unroll
      for (int ks = 0; ks < 2; ++ks)
        bk[t][ks] = *reinterpret_cast<const v16bf*>(kp + ks * 32 + 16 * hi);
    }
  };

  // One key block: consumes `cur` K fragments; preloads `nxt` for kb0+64.
  auto do_block = [&](int kb0, v16bf (&cur)[4][2], v16bf (&nxt)[4][2]) {
    // ---- scores: 8 back-to-back WMMAs ----
    v8f Sc[4] = {};
#pragma unroll
    for (int t = 0; t < 4; ++t)
#pragma unroll
      for (int ks = 0; ks < 2; ++ks)
        Sc[t] = wmma_bf16(aq[ks], cur[t][ks], Sc[t]);

    // ---- issue next block's K loads + this block's V loads; the softmax
    //      VALU work below hides both ----
    if (kb0 + 64 <= kend) load_bk(kb0 + 64, nxt);   // wave-uniform branch
    v16bf bv[2][4];
#pragma unroll
    for (int ks = 0; ks < 2; ++ks)
#pragma unroll
      for (int dt = 0; dt < 4; ++dt) {
        const __bf16* vp = Vh + (size_t)(dt * 16 + ln) * S_ + kb0 + ks * 32 + 16 * hi;
        bv[ks][dt] = *reinterpret_cast<const v16bf*>(vp);
      }

    // ---- scale + causal mask (branchless; -3e38 avoids inf-inf NaN) ----
#pragma unroll
    for (int t = 0; t < 4; ++t) {
      const int col = kb0 + t * 16 + ln;
#pragma unroll
      for (int r = 0; r < 8; ++r) {
        const int rowg = qb + r + 8 * hi;
        const float s = Sc[t][r] * 0.125f;            // 1/sqrt(64)
        Sc[t][r] = (col > rowg) ? -3.0e38f : s;
      }
    }

    // ---- per-row running max (reduce across 16-lane half groups) ----
    float mn[8], scale[8];
#pragma unroll
    for (int r = 0; r < 8; ++r) {
      float bm = fmaxf(fmaxf(Sc[0][r], Sc[1][r]), fmaxf(Sc[2][r], Sc[3][r]));
#pragma unroll
      for (int off = 1; off < 16; off <<= 1) bm = fmaxf(bm, __shfl_xor(bm, off, 32));
      mn[r] = fmaxf(m[r], bm);
      scale[r] = __expf(m[r] - mn[r]);
      m[r] = mn[r];
    }

    // ---- P = exp(S - m); row sums; stage P to LDS in [row][key] order ----
#pragma unroll
    for (int r = 0; r < 8; ++r) {
      float ls = 0.0f;
#pragma unroll
      for (int t = 0; t < 4; ++t) {
        const float p = __expf(Sc[t][r] - mn[r]);
        ls += p;
        Plds[wave][r + 8 * hi][t * 16 + ln] = (__bf16)p;
      }
#pragma unroll
      for (int off = 1; off < 16; off <<= 1) ls += __shfl_xor(ls, off, 32);
      l[r] = l[r] * scale[r] + ls;
    }

    // ---- rescale accumulator ----
#pragma unroll
    for (int t = 0; t < 4; ++t)
#pragma unroll
      for (int r = 0; r < 8; ++r) O[t][r] *= scale[r];

    // ---- O += P * V  (P via LDS transpose to A-layout; V already loaded) ----
#pragma unroll
    for (int ks = 0; ks < 2; ++ks) {
      v16bf ap = load_a_frag(&Plds[wave][ln][0], ks * 32, hi);
#pragma unroll
      for (int dt = 0; dt < 4; ++dt)
        O[dt] = wmma_bf16(ap, bv[ks][dt], O[dt]);
    }
  };

  // Unrolled-by-2 pipeline alternating the two K-fragment buffers
  // (avoids per-iteration register copies).
  v16bf bk0[4][2], bk1[4][2];
  load_bk(0, bk0);
  int kb0 = 0;
  for (;;) {
    do_block(kb0, bk0, bk1);
    kb0 += 64;
    if (kb0 > kend) break;
    do_block(kb0, bk1, bk0);
    kb0 += 64;
    if (kb0 > kend) break;
  }

  // ---- ghost softmax normalization: out = acc / (l + 1); store A bf16 ----
  const int b = bh >> 4, h = bh & (H_ - 1);
#pragma unroll
  for (int r = 0; r < 8; ++r) {
    const float inv = 1.0f / (l[r] + 1.0f);
    const int row = b * S_ + qb + r + 8 * hi;
#pragma unroll
    for (int dt = 0; dt < 4; ++dt)
      Abf[(size_t)row * (H_ * D_) + h * D_ + dt * 16 + ln] = (__bf16)(O[dt][r] * inv);
  }
}

// ---------------------------------------------------------------------------
// Stage 3: output projection.  A[BS][1024] * WoT[1024][1024] -> Out fp32.
// ---------------------------------------------------------------------------
__global__ __launch_bounds__(256) void gemm_out(const __bf16* __restrict__ A,
                                                const __bf16* __restrict__ Wt,
                                                float* __restrict__ Out) {
  const int lane = threadIdx.x & 31, wave = threadIdx.x >> 5;
  const int hi = lane >> 4, ln = lane & 15;
  const int wm = wave >> 2, wn = wave & 3;
  const int mbase = blockIdx.y * 64 + wm * 32;
  const int nbase = blockIdx.x * 256 + wn * 64;

  GemmCtx g = gemm_setup(A, Wt, mbase, nbase, ln, hi);
  v8f acc[2][4] = {};
  gemm_kloop(g, acc);

#pragma unroll
  for (int t = 0; t < 4; ++t) {
    const int col = nbase + t * 16 + ln;
#pragma unroll
    for (int mi = 0; mi < 2; ++mi)
#pragma unroll
      for (int r = 0; r < 8; ++r) {
        const int row = mbase + mi * 16 + r + 8 * hi;
        Out[(size_t)row * E_ + col] = acc[mi][t][r];
      }
  }
}

// ---------------------------------------------------------------------------
// Host-side orchestration (graph-capture safe: no mallocs/syncs).
// ---------------------------------------------------------------------------
extern "C" void kernel_launch(void* const* d_in, const int* in_sizes, int n_in,
                              void* d_out, int out_size, void* d_ws, size_t ws_size,
                              hipStream_t stream) {
  const float* X    = (const float*)d_in[0];   // (B,S,E)
  const float* Wqkv = (const float*)d_in[1];   // (E, 3*H*D)
  const float* Wo   = (const float*)d_in[2];   // (H*D, E)
  float* Out = (float*)d_out;                  // (B,S,E) fp32

  char* ws = (char*)d_ws;
  size_t off = 0;
  auto take = [&](size_t bytes) {
    char* p = ws + off;
    off += (bytes + 255) & ~(size_t)255;
    return p;
  };
  __bf16* Xbf    = (__bf16*)take((size_t)BS_ * E_ * 2);          // 8 MB
  __bf16* WqkvT  = (__bf16*)take((size_t)QKVN * E_ * 2);         // 6 MB
  __bf16* WoT    = (__bf16*)take((size_t)E_ * E_ * 2);           // 2 MB
  __bf16* Qb     = (__bf16*)take((size_t)B_ * H_ * S_ * D_ * 2); // 8 MB
  __bf16* Kb     = (__bf16*)take((size_t)B_ * H_ * S_ * D_ * 2); // 8 MB
  __bf16* Vt     = (__bf16*)take((size_t)B_ * H_ * D_ * S_ * 2); // 8 MB
  __bf16* Abf    = (__bf16*)take((size_t)BS_ * E_ * 2);          // 8 MB
  (void)ws_size; (void)in_sizes; (void)n_in; (void)out_size;

  // Stage 0: precision conversion + weight transposes.
  {
    int n = BS_ * E_;
    cast_bf16<<<(n + 255) / 256, 256, 0, stream>>>(X, Xbf, n);
  }
  transpose_bf16<<<dim3(QKVN / 32, E_ / 32), dim3(32, 8), 0, stream>>>(Wqkv, WqkvT, E_, QKVN);
  transpose_bf16<<<dim3(E_ / 32, E_ / 32), dim3(32, 8), 0, stream>>>(Wo, WoT, E_, E_);

  // Stage 1: QKV projection -> Q, K, V^T.
  gemm_qkv<<<dim3(QKVN / 256, BS_ / 64), 256, 0, stream>>>(Xbf, WqkvT, Qb, Kb, Vt);

  // Stage 2: causal flash attention with ghost softmax.
  attn_fwd<<<dim3(B_ * H_, S_ / 64), 128, 0, stream>>>(Qb, Kb, Vt, Abf);

  // Stage 3: output projection.
  gemm_out<<<dim3(E_ / 256, BS_ / 64), 256, 0, stream>>>(Abf, WoT, Out);
}
